// Model_8873402434274
// MI455X (gfx1250) — matrix operand order repaired and checked by reading, not compiled, not run on hardware
//
#include <hip/hip_runtime.h>
#include <hip/hip_bf16.h>
#include <math.h>

#define H_HEADS 6
#define CC 32
#define HC 192   // H*C
#define NEG_SLOPE 0.2f

typedef __attribute__((ext_vector_type(16))) _Float16 v16h;
typedef __attribute__((ext_vector_type(8)))  _Float16 v8h;
typedef __attribute__((ext_vector_type(8)))  float    v8f;

// ---------- gather i_emb = i_table[item] ----------
__global__ void k_gather_items(const int* __restrict__ item,
                               const float* __restrict__ itab,
                               float* __restrict__ x0, int B) {
    int t = blockIdx.x * blockDim.x + threadIdx.x;
    if (t >= B * 32) return;
    int n = t >> 5, c = t & 31;
    x0[t] = itab[item[n] * 32 + c];
}

// ---------- fp32 -> fp16 conversion ----------
__global__ void k_f32_to_f16(const float* __restrict__ in,
                             _Float16* __restrict__ out, int n) {
    int t = blockIdx.x * blockDim.x + threadIdx.x;
    if (t < n) out[t] = (_Float16)in[t];
}

// ---------- xh = X @ W  via V_WMMA_F32_16X16X32_F16 ----------
// X: [M,32] f16 row-major, W: [32,192] f16 row-major, XH: [M,192] f32
// One wave per 16-row M strip; the A operand is loaded once and reused
// across all 12 N tiles (12 WMMAs per wave). B (12 KB) stays L0-resident.
__global__ void k_gemm_wmma(const _Float16* __restrict__ X,
                            const _Float16* __restrict__ Wt,
                            float* __restrict__ XH, int M) {
    int gid  = blockIdx.x * blockDim.x + threadIdx.x;
    int mt   = gid >> 5;                      // wave id == M tile
    int lane = threadIdx.x & 31;
    int mtiles = (M + 15) >> 4;
    if (mt >= mtiles) return;                 // wave-uniform: EXEC all-ones below

    int half = lane >> 4;                     // 0: lanes 0-15, 1: lanes 16-31
    int lr   = lane & 15;

    // A (16x32 f16): lanes 0-15 = rows M0-15 holding K0-7,K16-23;
    //                lanes 16-31 = rows M0-15 holding K8-15,K24-31.
    int mrow  = mt * 16 + lr;
    int mload = (mrow < M) ? mrow : (M - 1);
    const _Float16* xrow = X + (size_t)mload * 32 + half * 8;
    v8h alo = *(const v8h*)(xrow);
    v8h ahi = *(const v8h*)(xrow + 16);
    v16h a;
#pragma unroll
    for (int i = 0; i < 8; ++i) { a[i] = alo[i]; a[8 + i] = ahi[i]; }

    const bool fullTile = (mt * 16 + 16) <= M;    // wave-uniform
    int rbase = mt * 16 + half * 8;

#pragma unroll
    for (int nt = 0; nt < 12; ++nt) {
        // B (32x16 f16): lane%16 = N column; halves hold K0-7,K16-23 (lanes 0-15) / K8-15,K24-31 (lanes 16-31).
        v16h b;
#pragma unroll
        for (int i = 0; i < 8; ++i) {
            b[i]     = Wt[(size_t)(half * 8 + i) * HC + nt * 16 + lr];
            b[8 + i] = Wt[(size_t)(16 + half * 8 + i) * HC + nt * 16 + lr];
        }
        v8f c = {};
        c = __builtin_amdgcn_wmma_f32_16x16x32_f16(
                /*neg_a=*/false, a, /*neg_b=*/false, b,
                /*c_mod=*/(short)0, c, /*reuse_a=*/false, /*reuse_b=*/false);

        // D: elem r -> row rbase + r, col nt*16 + lr
        int col = nt * 16 + lr;
        float* outp = XH + (size_t)rbase * HC + col;
        if (fullTile) {
#pragma unroll
            for (int r = 0; r < 8; ++r) outp[(size_t)r * HC] = c[r];
        } else {
#pragma unroll
            for (int r = 0; r < 8; ++r)
                if (rbase + r < M) outp[(size_t)r * HC] = c[r];
        }
    }
}

// ---------- per-(node,head) attention logit halves ----------
__global__ void k_alphas(const float* __restrict__ xh,
                         const float* __restrict__ a_src,
                         const float* __restrict__ a_dst,
                         float* __restrict__ as, float* __restrict__ ad, int B) {
    int t = blockIdx.x * blockDim.x + threadIdx.x;
    if (t >= B * H_HEADS) return;
    int n = t / H_HEADS, h = t % H_HEADS;
    const float* row = xh + (size_t)n * HC + h * CC;
    float s = 0.f, d = 0.f;
#pragma unroll
    for (int c = 0; c < CC; ++c) {
        float v = row[c];
        s += v * a_src[h * CC + c];
        d += v * a_dst[h * CC + c];
    }
    as[t] = s;
    ad[t] = d;
}

// ---------- init accumulator with bias, zero softmax denominators ----------
__global__ void k_init(const float* __restrict__ bias, float* __restrict__ y,
                       float* __restrict__ den, int B) {
    int t = blockIdx.x * blockDim.x + threadIdx.x;
    if (t >= B * 32) return;
    int c = t & 31;
    y[t] = bias[c];
    if (c < H_HEADS) den[(t >> 5) * H_HEADS + c] = 0.f;
}

// ---------- edge pass 1: e = exp(leaky_relu(as[src]+ad[dst])); denom[dst] += e ----------
__global__ void k_edge1(const int* __restrict__ g, int E, int B,
                        const float* __restrict__ as, const float* __restrict__ ad,
                        float* __restrict__ wedge, float* __restrict__ den) {
    int t = blockIdx.x * blockDim.x + threadIdx.x;
    int total = (E + B) * H_HEADS;
    if (t >= total) return;
    int e = t / H_HEADS, h = t % H_HEADS;
    int src, dst;
    if (e < E) { src = g[e]; dst = g[E + e]; } else { src = dst = e - E; }
    float v = as[src * H_HEADS + h] + ad[dst * H_HEADS + h];
    v = (v > 0.f) ? v : NEG_SLOPE * v;
    float ee = expf(v);
    wedge[t] = ee;
    atomicAdd(&den[dst * H_HEADS + h], ee);
}

// ---------- edge pass 2: y[dst,c] += (1/H) * sum_h alpha_eh * xh[src,h,c] ----------
// one wave per edge, lane = channel. Edge id / src / dst / weights / denoms are
// wave-uniform -> force to SGPRs so the backend uses scalar loads; only the
// coalesced xh gathers and the output atomics stay on the vector path.
__global__ void k_edge2(const int* __restrict__ g, int E, int B,
                        const float* __restrict__ wedge, const float* __restrict__ den,
                        const float* __restrict__ xh, float* __restrict__ y) {
    int gid  = blockIdx.x * blockDim.x + threadIdx.x;
    int e    = gid >> 5;
    int lane = threadIdx.x & 31;
    if (e >= E + B) return;                       // wave-uniform
    e = __builtin_amdgcn_readfirstlane(e);
    int src, dst;
    if (e < E) { src = g[e]; dst = g[E + e]; } else { src = dst = e - E; }
    src = __builtin_amdgcn_readfirstlane(src);
    dst = __builtin_amdgcn_readfirstlane(dst);
    float acc = 0.f;
#pragma unroll
    for (int h = 0; h < H_HEADS; ++h) {
        float w = wedge[e * H_HEADS + h];
        float d = den[dst * H_HEADS + h];
        float alpha = w / (d + 1e-16f);
        acc += alpha * xh[(size_t)src * HC + h * CC + lane];
    }
    atomicAdd(&y[dst * 32 + lane], acc * (1.0f / (float)H_HEADS));
}

// ---------- final MLP: sigmoid((concat(u,i) @ lw1 + lb1) @ lw2 + lb2) ----------
// 4 rows per 256-thread block; 64 threads per row (one per hidden unit)
__global__ void k_mlp(const int* __restrict__ user, const float* __restrict__ utab,
                      const float* __restrict__ y2,
                      const float* __restrict__ lw1, const float* __restrict__ lb1,
                      const float* __restrict__ lw2, const float* __restrict__ lb2,
                      float* __restrict__ out, int B) {
    __shared__ float sin_[4][64];
    __shared__ float spart[4][64];
    int r = threadIdx.x >> 6;
    int j = threadIdx.x & 63;
    int row = blockIdx.x * 4 + r;
    if (row < B) {
        float v;
        if (j < 32) v = utab[user[row] * 32 + j];
        else        v = y2[row * 32 + (j - 32)];
        sin_[r][j] = v;
    }
    __syncthreads();
    if (row < B) {
        float hsum = lb1[j];
#pragma unroll
        for (int k = 0; k < 64; ++k) hsum += sin_[r][k] * lw1[k * 64 + j];
        spart[r][j] = hsum * lw2[j];
    }
    __syncthreads();
    if (row < B && j == 0) {
        float s = lb2[0];
        for (int k = 0; k < 64; ++k) s += spart[r][k];
        out[row] = 1.f / (1.f + expf(-s));
    }
}

extern "C" void kernel_launch(void* const* d_in, const int* in_sizes, int n_in,
                              void* d_out, int out_size, void* d_ws, size_t ws_size,
                              hipStream_t stream) {
    (void)n_in; (void)out_size; (void)ws_size;
    const int*   user = (const int*)d_in[0];
    const int*   item = (const int*)d_in[1];
    const int*   graph= (const int*)d_in[2];
    const float* utab = (const float*)d_in[3];
    const float* itab = (const float*)d_in[4];
    const float* W1   = (const float*)d_in[5];
    const float* asr1 = (const float*)d_in[6];
    const float* ads1 = (const float*)d_in[7];
    const float* b1   = (const float*)d_in[8];
    const float* W2   = (const float*)d_in[9];
    const float* asr2 = (const float*)d_in[10];
    const float* ads2 = (const float*)d_in[11];
    const float* b2   = (const float*)d_in[12];
    const float* lw1  = (const float*)d_in[13];
    const float* lb1  = (const float*)d_in[14];
    const float* lw2  = (const float*)d_in[15];
    const float* lb2  = (const float*)d_in[16];

    const int B = in_sizes[0];
    const int E = in_sizes[2] / 2;

    // workspace carve-out (all fits in L2: ~85 MB)
    char* ws = (char*)d_ws;
    size_t off = 0;
    auto alloc = [&](size_t bytes) -> void* {
        void* p = ws + off;
        off += (bytes + 255) & ~(size_t)255;
        return p;
    };
    float*    x0    = (float*)   alloc((size_t)B * 32 * 4);
    float*    y1    = (float*)   alloc((size_t)B * 32 * 4);
    float*    y2    = (float*)   alloc((size_t)B * 32 * 4);
    _Float16* x16   = (_Float16*)alloc((size_t)B * 32 * 2);
    _Float16* w16   = (_Float16*)alloc((size_t)32 * HC * 2);
    float*    xh    = (float*)   alloc((size_t)B * HC * 4);
    float*    asb   = (float*)   alloc((size_t)B * H_HEADS * 4);
    float*    adb   = (float*)   alloc((size_t)B * H_HEADS * 4);
    float*    den   = (float*)   alloc((size_t)B * H_HEADS * 4);
    float*    wedge = (float*)   alloc((size_t)(E + B) * H_HEADS * 4);

    const int TB = 256;
    auto grid = [&](long long n) { return (unsigned)((n + TB - 1) / TB); };

    k_gather_items<<<grid((long long)B * 32), TB, 0, stream>>>(item, itab, x0, B);

    const float* xin = x0;
    float* yout = y1;
    for (int layer = 0; layer < 2; ++layer) {
        const float* Wl  = layer ? W2   : W1;
        const float* asl = layer ? asr2 : asr1;
        const float* adl = layer ? ads2 : ads1;
        const float* bl  = layer ? b2   : b1;

        k_f32_to_f16<<<grid((long long)B * 32), TB, 0, stream>>>(xin, x16, B * 32);
        k_f32_to_f16<<<grid(32 * HC), TB, 0, stream>>>(Wl, w16, 32 * HC);

        long long waves = (long long)((B + 15) / 16);   // one wave per 16-row strip
        k_gemm_wmma<<<grid(waves * 32), TB, 0, stream>>>(x16, w16, xh, B);

        k_alphas<<<grid((long long)B * H_HEADS), TB, 0, stream>>>(xh, asl, adl, asb, adb, B);
        k_init<<<grid((long long)B * 32), TB, 0, stream>>>(bl, yout, den, B);
        k_edge1<<<grid((long long)(E + B) * H_HEADS), TB, 0, stream>>>(graph, E, B, asb, adb, wedge, den);
        k_edge2<<<grid((long long)(E + B) * 32), TB, 0, stream>>>(graph, E, B, wedge, den, xh, yout);

        xin  = yout;   // layer 2 consumes y1
        yout = y2;     // layer 2 writes y2
    }

    k_mlp<<<(B + 3) / 4, 256, 0, stream>>>(user, utab, y2, lw1, lb1, lw2, lb2,
                                           (float*)d_out, B);
}
